// MGIN_84361747628355
// MI455X (gfx1250) — matrix operand order repaired and checked by reading, not compiled, not run on hardware
//
#include <hip/hip_runtime.h>

typedef float v8f __attribute__((ext_vector_type(8)));
typedef __bf16 v16bf __attribute__((ext_vector_type(16)));

union BFPack { uint4 u[2]; v16bf v; };

constexpr int NNODE = 384;
constexpr int NE    = 12288;
constexpr int DD    = 1280;
constexpr int HH    = 128;
constexpr long NPAIR = (long)NNODE * NNODE;   // 147456

#define SQB(A, Cc) ((__bf16)(((A) - (Cc)) * ((A) - (Cc))))

// ---------------- setup kernels ----------------
__global__ void k_build_x0(const float* __restrict__ lm, const float* __restrict__ nf,
                           float* __restrict__ x0) {
  int idx = blockIdx.x * 256 + threadIdx.x;
  if (idx >= NNODE * DD) return;
  int v = idx / DD, d = idx % DD;
  x0[idx] = (d < 1024) ? lm[(size_t)(v + 1) * 1024 + d]
                       : nf[(size_t)v * 256 + (d - 1024)];
}

// transpose fp32 [K][Nc] -> bf16 [Nc][K]
__global__ void k_transb(const float* __restrict__ W, __bf16* __restrict__ WT,
                         int K, int Nc) {
  int idx = blockIdx.x * 256 + threadIdx.x;
  if (idx >= K * Nc) return;
  int n = idx / K, k = idx % K;
  WT[idx] = (__bf16)W[(size_t)k * Nc + n];
}

__global__ void k_wt2(const float* __restrict__ Wd2, __bf16* __restrict__ WT2) {
  int idx = blockIdx.x * 256 + threadIdx.x;            // [n][k], n<32 (pad 30->32), k<128
  if (idx >= 32 * HH) return;
  int n = idx / HH, k = idx % HH;
  WT2[idx] = (n < 30) ? (__bf16)Wd2[(size_t)k * 30 + n] : (__bf16)0.f;
}

// ---------------- deterministic CSR build ----------------
__global__ void k_count(const int* __restrict__ dst, int* __restrict__ cnt) {
  int e = blockIdx.x * 256 + threadIdx.x;
  if (e < NE) atomicAdd(&cnt[dst[e]], 1);
}
__global__ void k_offsets(const int* __restrict__ cnt, int* __restrict__ offs) {
  if (threadIdx.x == 0 && blockIdx.x == 0) {
    int s = 0;
    for (int v = 0; v < NNODE; ++v) { offs[v] = s; s += cnt[v]; }
    offs[NNODE] = s;
  }
}
__global__ void k_fill(const int* __restrict__ dst, const int* __restrict__ offs,
                       int* __restrict__ cursor, int* __restrict__ elist) {
  int e = blockIdx.x * 256 + threadIdx.x;
  if (e < NE) {
    int d = dst[e];
    int p = atomicAdd(&cursor[d], 1);
    elist[offs[d] + p] = e;
  }
}
__global__ void k_sort(const int* __restrict__ offs, int* __restrict__ elist) {
  int v = blockIdx.x * blockDim.x + threadIdx.x;
  if (v >= NNODE) return;
  int b = offs[v], e = offs[v + 1];
  for (int a = b + 1; a < e; ++a) {
    int key = elist[a]; int q = a - 1;
    while (q >= b && elist[q] > key) { elist[q + 1] = elist[q]; --q; }
    elist[q + 1] = key;
  }
}

// y[v,:] = bf16( x[v,:] + sum_{e: dst==v} w_e * x[src_e,:] )  (fixed order -> deterministic)
__global__ __launch_bounds__(256)
void k_agg(const float* __restrict__ xin, __bf16* __restrict__ yout,
           const int* __restrict__ offs, const int* __restrict__ elist,
           const float* __restrict__ edge_feat, const int* __restrict__ src) {
  int v = blockIdx.x, t = threadIdx.x;
  float acc[5] = {0.f, 0.f, 0.f, 0.f, 0.f};
  int b = offs[v], e = offs[v + 1];
  for (int idx = b; idx < e; ++idx) {
    int ed = elist[idx];
    float ef = edge_feat[ed];
    float w = 1.0f / (ef * ef + 1e-6f);
    const float* xs = xin + (size_t)src[ed] * DD;
    #pragma unroll
    for (int j = 0; j < 5; ++j) acc[j] += w * xs[t + 256 * j];
  }
  const float* xv = xin + (size_t)v * DD;
  #pragma unroll
  for (int j = 0; j < 5; ++j)
    yout[(size_t)v * DD + t + 256 * j] = (__bf16)(xv[t + 256 * j] + acc[j]);
}

// ---------------- GIN GEMM: out = Ybf @ W + bias (+ res), bf16 WMMA ----------------
// Ybf: bf16 [384][1280] row-major; WTg: bf16 [n][k] (pre-transposed)
__global__ __launch_bounds__(128)
void k_gin(const __bf16* __restrict__ Ybf, const __bf16* __restrict__ WTg,
           const float* __restrict__ bias, const float* __restrict__ res,
           float* __restrict__ out) {
  __shared__ __align__(16) __bf16 lds_a[64 * 40];
  __shared__ __align__(16) __bf16 lds_w[64 * 40];
  const int t = threadIdx.x, wv = t >> 5, l = t & 31, ml = l & 15, hi = l >> 4;
  const int m0 = blockIdx.x * 64, n0 = blockIdx.y * 64;
  const int sr = t >> 1, hp = t & 1;      // staging: row 0..63, 16-elem half

  v8f acc[4];
  #pragma unroll
  for (int c = 0; c < 4; ++c) {
    float bv = bias[n0 + 16 * c + ml];
    #pragma unroll
    for (int r = 0; r < 8; ++r) acc[c][r] = bv;
  }

  for (int k0 = 0; k0 < DD; k0 += 32) {
    __syncthreads();
    { // A tile 64x32 (uint4 copies)
      const uint4* s = (const uint4*)(Ybf + (size_t)(m0 + sr) * DD + k0 + 16 * hp);
      uint4* dp = (uint4*)(lds_a + sr * 40 + 16 * hp);
      dp[0] = s[0]; dp[1] = s[1];
    }
    { // W tile [n][k] 64x32
      const uint4* s = (const uint4*)(WTg + (size_t)(n0 + sr) * DD + k0 + 16 * hp);
      uint4* dp = (uint4*)(lds_w + sr * 40 + 16 * hp);
      dp[0] = s[0]; dp[1] = s[1];
    }
    __syncthreads();

    int row = 16 * wv + ml;
    BFPack ap;
    ap.u[0] = *(const uint4*)(lds_a + row * 40 + 8 * hi);
    ap.u[1] = *(const uint4*)(lds_a + row * 40 + 16 + 8 * hi);
    BFPack bp[4];
    #pragma unroll
    for (int c = 0; c < 4; ++c) {
      bp[c].u[0] = *(const uint4*)(lds_w + (16 * c + ml) * 40 + 8 * hi);
      bp[c].u[1] = *(const uint4*)(lds_w + (16 * c + ml) * 40 + 16 + 8 * hi);
    }
    #pragma unroll
    for (int c = 0; c < 4; ++c)
      acc[c] = __builtin_amdgcn_wmma_f32_16x16x32_bf16(
          false, ap.v, false, bp[c].v, (short)0, acc[c], false, false);
  }

  #pragma unroll
  for (int c = 0; c < 4; ++c) {
    int colg = n0 + 16 * c + ml;
    #pragma unroll
    for (int r = 0; r < 8; ++r) {
      int rowg = m0 + 16 * wv + r + 8 * hi;
      float v = acc[c][r];
      if (res) v += res[(size_t)rowg * DD + colg];
      out[(size_t)rowg * DD + colg] = v;
    }
  }
}

// ---------------- fused pairwise dis_nn kernel ----------------
// block: 128 pair-rows (fixed i) x 128 hidden cols; 8 waves, 8 col tiles each.
// LDS overlay: [ w_buf0 | w_buf1 | x_i ] reused as [ hidden ] after K-loop; w2 persistent.
__global__ __launch_bounds__(256)
void k_dis(const float* __restrict__ x, const __bf16* __restrict__ WT1,
           const float* __restrict__ bd1, const __bf16* __restrict__ WT2,
           const float* __restrict__ bd2, float* __restrict__ out) {
  __shared__ __align__(16) char smem[34816 + 8704];
  __bf16* lds_wb[2];
  lds_wb[0]       = (__bf16*)smem;                  // 10240 B
  lds_wb[1]       = (__bf16*)(smem + 10240);        // 10240 B
  float*  lds_xi  = (float*)(smem + 20480);         // 5120 B
  __bf16* lds_hid = (__bf16*)smem;                  // 34816 B (after K-loop)
  __bf16* lds_w2  = (__bf16*)(smem + 34816);        // 8704 B (persistent)

  const int t = threadIdx.x;
  const int wv = t >> 5, l = t & 31, ml = l & 15, hi = l >> 4;
  const int sn = t >> 1, sp = t & 1;                // staging: n-row, 16-elem half
  const long row0 = (long)blockIdx.x * 128;
  const int i  = (int)(row0 / NNODE);
  const int j0 = (int)(row0 % NNODE);

  { // stage x_i row, Wd2^T, and first Wd1 K-tile
    const float4* xi4 = (const float4*)(x + (size_t)i * DD);
    for (int idx = t; idx < DD / 4; idx += 256) ((float4*)lds_xi)[idx] = xi4[idx];
    int n2 = t >> 3, h2 = t & 7;
    const uint4* s2 = (const uint4*)(WT2 + n2 * HH + 16 * h2);
    uint4* d2 = (uint4*)(lds_w2 + n2 * 136 + 16 * h2);
    d2[0] = s2[0]; d2[1] = s2[1];
    const uint4* s = (const uint4*)(WT1 + (size_t)sn * DD + 16 * sp);
    uint4* dp = (uint4*)(lds_wb[0] + sn * 40 + 16 * sp);
    dp[0] = s[0]; dp[1] = s[1];
  }
  __syncthreads();

  v8f acc[8];
  #pragma unroll
  for (int c = 0; c < 8; ++c) {
    float bv = bd1[16 * c + ml];
    #pragma unroll
    for (int r = 0; r < 8; ++r) acc[c][r] = bv;
  }

  const int j = j0 + 16 * wv + ml;
  const float* xj = x + (size_t)j * DD;

  for (int k0 = 0; k0 < DD; k0 += 32) {
    const __bf16* cur = lds_wb[(k0 >> 5) & 1];
    if (k0 + 32 < DD) {                    // stage next tile into other buffer
      __bf16* nxt = lds_wb[((k0 >> 5) + 1) & 1];
      const uint4* s = (const uint4*)(WT1 + (size_t)sn * DD + (k0 + 32) + 16 * sp);
      uint4* dp = (uint4*)(nxt + sn * 40 + 16 * sp);
      dp[0] = s[0]; dp[1] = s[1];
      __builtin_prefetch(xj + k0 + 32, 0, 1);
    }

    // A fragment = bf16((x_j - x_i)^2), 16x32 layout
    const float4* pa  = (const float4*)(xj + k0 + 8 * hi);
    const float4* pa2 = (const float4*)(xj + k0 + 16 + 8 * hi);
    float4 a0 = pa[0],  a1 = pa[1];
    float4 a2 = pa2[0], a3 = pa2[1];
    const float4* pc  = (const float4*)(lds_xi + k0 + 8 * hi);
    const float4* pc2 = (const float4*)(lds_xi + k0 + 16 + 8 * hi);
    float4 c0 = pc[0],  c1 = pc[1];
    float4 c2 = pc2[0], c3 = pc2[1];
    v16bf af;
    af[0]  = SQB(a0.x, c0.x); af[1]  = SQB(a0.y, c0.y);
    af[2]  = SQB(a0.z, c0.z); af[3]  = SQB(a0.w, c0.w);
    af[4]  = SQB(a1.x, c1.x); af[5]  = SQB(a1.y, c1.y);
    af[6]  = SQB(a1.z, c1.z); af[7]  = SQB(a1.w, c1.w);
    af[8]  = SQB(a2.x, c2.x); af[9]  = SQB(a2.y, c2.y);
    af[10] = SQB(a2.z, c2.z); af[11] = SQB(a2.w, c2.w);
    af[12] = SQB(a3.x, c3.x); af[13] = SQB(a3.y, c3.y);
    af[14] = SQB(a3.z, c3.z); af[15] = SQB(a3.w, c3.w);

    // group loads (4 fragments) then group WMMAs for denser matrix-pipe issue
    BFPack bp[4];
    #pragma unroll
    for (int c = 0; c < 4; ++c) {
      bp[c].u[0] = *(const uint4*)(cur + (16 * c + ml) * 40 + 8 * hi);
      bp[c].u[1] = *(const uint4*)(cur + (16 * c + ml) * 40 + 16 + 8 * hi);
    }
    #pragma unroll
    for (int c = 0; c < 4; ++c)
      acc[c] = __builtin_amdgcn_wmma_f32_16x16x32_bf16(
          false, af, false, bp[c].v, (short)0, acc[c], false, false);
    #pragma unroll
    for (int c = 0; c < 4; ++c) {
      bp[c].u[0] = *(const uint4*)(cur + (16 * (c + 4) + ml) * 40 + 8 * hi);
      bp[c].u[1] = *(const uint4*)(cur + (16 * (c + 4) + ml) * 40 + 16 + 8 * hi);
    }
    #pragma unroll
    for (int c = 0; c < 4; ++c)
      acc[c + 4] = __builtin_amdgcn_wmma_f32_16x16x32_bf16(
          false, af, false, bp[c].v, (short)0, acc[c + 4], false, false);

    __syncthreads();   // reads of cur done; writes to nxt done
  }

  // stage 2: relu -> bf16 hidden in (wave-private) LDS, overlaying dead buffers
  #pragma unroll
  for (int c = 0; c < 8; ++c) {
    #pragma unroll
    for (int r = 0; r < 8; ++r) {
      float v = acc[c][r];
      v = v > 0.f ? v : 0.f;
      lds_hid[(16 * wv + r + 8 * hi) * 136 + 16 * c + ml] = (__bf16)v;
    }
  }

  v8f acc2[2];
  #pragma unroll
  for (int c = 0; c < 2; ++c) {
    int col = 16 * c + ml;
    float bv = (col < 30) ? bd2[col] : 0.f;
    #pragma unroll
    for (int r = 0; r < 8; ++r) acc2[c][r] = bv;
  }
  #pragma unroll
  for (int kk = 0; kk < 4; ++kk) {
    BFPack ap;
    ap.u[0] = *(const uint4*)(lds_hid + (16 * wv + ml) * 136 + 32 * kk + 8 * hi);
    ap.u[1] = *(const uint4*)(lds_hid + (16 * wv + ml) * 136 + 32 * kk + 16 + 8 * hi);
    BFPack bq[2];
    #pragma unroll
    for (int c = 0; c < 2; ++c) {
      bq[c].u[0] = *(const uint4*)(lds_w2 + (16 * c + ml) * 136 + 32 * kk + 8 * hi);
      bq[c].u[1] = *(const uint4*)(lds_w2 + (16 * c + ml) * 136 + 32 * kk + 16 + 8 * hi);
    }
    #pragma unroll
    for (int c = 0; c < 2; ++c)
      acc2[c] = __builtin_amdgcn_wmma_f32_16x16x32_bf16(
          false, ap.v, false, bq[c].v, (short)0, acc2[c], false, false);
  }

  #pragma unroll
  for (int c = 0; c < 2; ++c) {
    int col = 16 * c + ml;
    if (col < 30) {
      #pragma unroll
      for (int r = 0; r < 8; ++r) {
        long rowg = row0 + 16 * wv + r + 8 * hi;
        out[rowg * 30 + col] = acc2[c][r];
      }
    }
  }
}

// ---------------- mask_nn (tiny) ----------------
__global__ __launch_bounds__(128)
void k_mask(const float* __restrict__ x, const int* __restrict__ midx,
            const float* __restrict__ Wm1, const float* __restrict__ bm1,
            const float* __restrict__ Wm2, const float* __restrict__ bm2,
            float* __restrict__ outm) {
  __shared__ float hid[HH];
  int m = blockIdx.x, h = threadIdx.x;
  const float* xr = x + (size_t)midx[m] * DD;
  float a = bm1[h];
  for (int d = 0; d < DD; ++d) a += xr[d] * Wm1[(size_t)d * HH + h];
  hid[h] = a > 0.f ? a : 0.f;
  __syncthreads();
  if (h < 2) {
    float a2 = bm2[h];
    for (int q = 0; q < HH; ++q) a2 += hid[q] * Wm2[(size_t)q * 2 + h];
    outm[(size_t)m * 2 + h] = tanhf(a2);
  }
}

extern "C" void kernel_launch(void* const* d_in, const int* in_sizes, int n_in,
                              void* d_out, int out_size, void* d_ws, size_t ws_size,
                              hipStream_t stream) {
  (void)in_sizes; (void)n_in; (void)out_size; (void)ws_size;
  const float* lm   = (const float*)d_in[0];
  const float* nf   = (const float*)d_in[1];
  const float* ef   = (const float*)d_in[2];
  const int*   src  = (const int*)d_in[3];
  const int*   dst  = (const int*)d_in[4];
  const int*   midx = (const int*)d_in[5];
  const float* Wg0  = (const float*)d_in[6];
  const float* bg0  = (const float*)d_in[7];
  const float* Wg1  = (const float*)d_in[8];
  const float* bg1  = (const float*)d_in[9];
  const float* Wd1  = (const float*)d_in[10];
  const float* bd1  = (const float*)d_in[11];
  const float* Wd2  = (const float*)d_in[12];
  const float* bd2  = (const float*)d_in[13];
  const float* Wm1  = (const float*)d_in[14];
  const float* bm1  = (const float*)d_in[15];
  const float* Wm2  = (const float*)d_in[16];
  const float* bm2  = (const float*)d_in[17];

  float* x0 = (float*)d_ws;                      // fp32 [384][1280]
  float* hb = x0 + (size_t)NNODE * DD;           // fp32 GIN hidden
  float* xb = hb + (size_t)NNODE * DD;           // fp32 node_output
  __bf16* ybf  = (__bf16*)(xb + (size_t)NNODE * DD);   // bf16 agg output (reused)
  __bf16* WT1  = ybf + (size_t)NNODE * DD;             // bf16 Wd1^T [128][1280]
  __bf16* WT2  = WT1 + (size_t)HH * DD;                // bf16 Wd2^T padded [32][128]
  __bf16* WTg0 = WT2 + 32 * HH;                        // bf16 Wgin0^T [1280][1280]
  __bf16* WTg1 = WTg0 + (size_t)DD * DD;
  int* cnt    = (int*)(WTg1 + (size_t)DD * DD);
  int* offs   = cnt + NNODE;
  int* cursor = offs + (NNODE + 16);
  int* elist  = cursor + NNODE;

  hipMemsetAsync(cnt, 0, NNODE * sizeof(int), stream);
  hipMemsetAsync(cursor, 0, NNODE * sizeof(int), stream);

  k_build_x0<<<(NNODE * DD + 255) / 256, 256, 0, stream>>>(lm, nf, x0);
  k_transb<<<(HH * DD + 255) / 256, 256, 0, stream>>>(Wd1, WT1, DD, HH);
  k_wt2<<<(32 * HH + 255) / 256, 256, 0, stream>>>(Wd2, WT2);
  k_transb<<<(DD * DD + 255) / 256, 256, 0, stream>>>(Wg0, WTg0, DD, DD);
  k_transb<<<(DD * DD + 255) / 256, 256, 0, stream>>>(Wg1, WTg1, DD, DD);

  k_count<<<NE / 256, 256, 0, stream>>>(dst, cnt);
  k_offsets<<<1, 1, 0, stream>>>(cnt, offs);
  k_fill<<<NE / 256, 256, 0, stream>>>(dst, offs, cursor, elist);
  k_sort<<<3, 128, 0, stream>>>(offs, elist);

  k_agg<<<NNODE, 256, 0, stream>>>(x0, ybf, offs, elist, ef, src);
  k_gin<<<dim3(6, 20), 128, 0, stream>>>(ybf, WTg0, bg0, nullptr, hb);
  k_agg<<<NNODE, 256, 0, stream>>>(hb, ybf, offs, elist, ef, src);
  k_gin<<<dim3(6, 20), 128, 0, stream>>>(ybf, WTg1, bg1, x0, xb);

  float* out = (float*)d_out;
  k_dis<<<(int)(NPAIR / 128), 256, 0, stream>>>(xb, WT1, bd1, WT2, bd2, out);
  k_mask<<<64, 128, 0, stream>>>(xb, midx, Wm1, bm1, Wm2, bm2, out + NPAIR * 30);
}
